// ConformerEnsembleEmbeddingCombiner_1271310320427
// MI455X (gfx1250) — compile-verified, matrix operand order checked by reading, not run on hardware
//
#include <hip/hip_runtime.h>
#include <hip/hip_bf16.h>

typedef __bf16 bf16;
typedef __attribute__((ext_vector_type(16))) __bf16 v16bf;
typedef __attribute__((ext_vector_type(8)))  __bf16 v8bf;
typedef __attribute__((ext_vector_type(8)))  float   v8f;
typedef __attribute__((ext_vector_type(4)))  unsigned int v4u;
typedef __attribute__((ext_vector_type(8)))  int v8i;
typedef __attribute__((ext_vector_type(4)))  int v4i;

#define NTOK 65536
#define GRP  32
#define NB   2048
#define SD   256
#define VD   768
#define DD   1024

__device__ __forceinline__ float silu_f(float x) {
    return x / (1.f + __expf(-x));
}

// Build a 16x32 bf16 WMMA fragment for this lane from a row of 32 contiguous
// K-values (row-of-A or column-of-W layout): lane l (row/col = l&15, khalf =
// l>>4) holds K = khalf*8 + [0..7] and 16 + khalf*8 + [0..7].  Two 16-byte
// aligned loads (ds_load_b128 from LDS, global_load_b128 from memory).
__device__ __forceinline__ v16bf mk_frag(const bf16* __restrict__ p) {
    v8bf lo = *(const v8bf*)p;
    v8bf hi = *(const v8bf*)(p + 16);
    v16bf f;
#pragma unroll
    for (int e = 0; e < 8; e++) { f[e] = lo[e]; f[e + 8] = hi[e]; }
    return f;
}

// ---- Tensor Data Mover: async DMA of a 2-D bf16 tile (rows x cols) from
// global memory (row stride = stride_elems) into LDS at lds_off, with the
// hardware LDS-padding feature inserting 16 B after every 128 B row so the
// LDS row stride becomes 144 B (72 bf16) — matching the compute-side layout.
// Descriptor per CDNA5 ISA ch.8: group0 {count=1, lds_addr, global_addr,
// type=2}; group1 {data_size=2B, pad_enable, pad_interval=32dw, pad_amount=4dw,
// tensor_dim0=cols, tensor_dim1=rows, tile_dim0=cols, tile_dim1=rows,
// tensor_dim0_stride=stride}.  Tracked with TENSORcnt.
__device__ __forceinline__ void tdm_load_2d(unsigned int lds_off,
                                            const void* gptr,
                                            unsigned int rows,
                                            unsigned int cols,
                                            unsigned int stride_elems) {
    unsigned long long ga = (unsigned long long)(uintptr_t)gptr;
    v4u g0;
    g0[0] = 1u;                                   // count=1 (valid user D#)
    g0[1] = lds_off;                              // LDS byte address
    g0[2] = (unsigned int)ga;                     // global_addr[31:0]
    g0[3] = (unsigned int)(ga >> 32) | (2u << 30);// global_addr[56:32] | type=2
    v8i g1;
    g1[0] = (1 << 16)    // data_size = 2 bytes
          | (1 << 20)    // pad_enable
          | (4 << 22)    // pad_interval: 2^(4+1) = 32 DWORDs = 128 B
          | (3 << 25);   // pad_amount: 3+1 = 4 DWORDs = 16 B
    g1[1] = (int)(cols << 16);          // tensor_dim0[15:0] (bits 63:48)
    g1[2] = (int)(rows << 16);          // tensor_dim0 hi=0, tensor_dim1[15:0]
    g1[3] = (int)(cols << 16);          // tensor_dim1 hi=0, tile_dim0
    g1[4] = (int)rows;                  // tile_dim1 (tile_dim2=0)
    g1[5] = (int)stride_elems;          // tensor_dim0_stride[31:0]
    g1[6] = 0;                          // stride hi, tensor_dim1_stride lo
    g1[7] = 0;
    v4i gz; gz[0] = gz[1] = gz[2] = gz[3] = 0;
#if defined(__clang_major__) && __clang_major__ >= 23
    v8i gz8;
#pragma unroll
    for (int e = 0; e < 8; e++) gz8[e] = 0;
    __builtin_amdgcn_tensor_load_to_lds(g0, g1, gz, gz, gz8, 0);
#else
    __builtin_amdgcn_tensor_load_to_lds(g0, g1, gz, gz, 0);
#endif
}

// ---------------- weight transpose + fp32->bf16 convert ----------------
// w: K x Nc (row major, as used by  x @ w)  ->  wt: Nc x K (row major)
__global__ void k_conv_wt(const float* __restrict__ w, bf16* __restrict__ wt,
                          int K, int Nc) {
    int idx = blockIdx.x * blockDim.x + threadIdx.x;
    if (idx >= K * Nc) return;
    int n = idx / K, k = idx - n * K;
    wt[(size_t)n * K + k] = (bf16)w[(size_t)k * Nc + n];
}

// ---------------- exact fp32 group means of the input slices ----------------
__global__ void k_meanpool_input(const float* __restrict__ emb,
                                 float* __restrict__ out_s,
                                 float* __restrict__ out_v) {
    int idx = blockIdx.x * blockDim.x + threadIdx.x;
    if (idx >= NB * DD) return;
    int b = idx >> 10, d = idx & (DD - 1);
    const float* p = emb + (size_t)b * GRP * DD + d;
    float s = 0.f;
#pragma unroll
    for (int g = 0; g < GRP; g++) s += p[(size_t)g * DD];
    s *= (1.f / GRP);
    if (d < SD) out_s[(size_t)b * SD + d] = s;
    else        out_v[(size_t)b * VD + (d - SD)] = s;
}

// ---------------- bf16 group mean (f32 accumulate) ----------------
__global__ void k_pool_bf16(const bf16* __restrict__ in, bf16* __restrict__ out,
                            int C) {
    int idx = blockIdx.x * blockDim.x + threadIdx.x;
    if (idx >= NB * C) return;
    int b = idx / C, c = idx - b * C;
    const bf16* p = in + (size_t)b * GRP * C + c;
    float s = 0.f;
#pragma unroll
    for (int g = 0; g < GRP; g++) s += (float)p[(size_t)g * C];
    out[(size_t)b * C + c] = (bf16)(s * (1.f / GRP));
}

// ---- shared compute core: 2 WMMA K-steps on a staged 128x(64+pad) tile pair
__device__ __forceinline__ void wmma_block(const bf16* as, const bf16* bs,
                                           int wm, int wn, int lr, int kh,
                                           v8f acc[4][2]) {
    constexpr int LDSS = 72;
#pragma unroll
    for (int ks = 0; ks < 2; ks++) {
        v16bf af[4], bfv[2];
#pragma unroll
        for (int i = 0; i < 4; i++)
            af[i] = mk_frag(&as[(wm + i * 16 + lr) * LDSS + ks * 32 + kh * 8]);
#pragma unroll
        for (int j = 0; j < 2; j++)
            bfv[j] = mk_frag(&bs[(wn + j * 16 + lr) * LDSS + ks * 32 + kh * 8]);
#pragma unroll
        for (int i = 0; i < 4; i++)
#pragma unroll
            for (int j = 0; j < 2; j++)
                acc[i][j] = __builtin_amdgcn_wmma_f32_16x16x32_bf16(
                    false, af[i], false, bfv[j], (short)0, acc[i][j], false, false);
    }
}

// ---- shared epilogue: bias + optional SiLU, scatter C tiles ----
template <typename OT, bool SILU>
__device__ __forceinline__ void gemm_epilogue(v8f acc[4][2],
                                              const float* __restrict__ bias,
                                              OT* __restrict__ out, int ldo,
                                              int m0, int n0, int wm, int wn,
                                              int lr, int kh) {
#pragma unroll
    for (int i = 0; i < 4; i++)
#pragma unroll
        for (int j = 0; j < 2; j++) {
            const int n = n0 + wn + j * 16 + lr;
            const float bv = bias[n];
            const int mb = m0 + wm + i * 16 + kh * 8;
#pragma unroll
            for (int r = 0; r < 8; r++) {
                float v = acc[i][j][r] + bv;
                if constexpr (SILU) v = silu_f(v);
                out[(size_t)(mb + r) * ldo + n] = (OT)v;
            }
        }
}

// ---------------- fp32-input WMMA GEMM (VALU-converted staging) ------------
// A: M x K fp32 (row stride lda, converted to bf16 while staging to LDS)
// Wt: Nc x K bf16 row-major.  Block tile 128x128x64, 8 waves (2M x 4N).
template <typename OT, bool SILU>
__global__ __launch_bounds__(256)
void k_gemm_f32(const float* __restrict__ A, int lda,
                const bf16* __restrict__ Wt,
                const float* __restrict__ bias,
                OT* __restrict__ out, int ldo, int K) {
    constexpr int BM = 128, BN = 128, BK = 64, LDSS = BK + 8;
    __shared__ __align__(16) bf16 As[BM * LDSS];
    __shared__ __align__(16) bf16 Bs[BN * LDSS];

    const int tid  = threadIdx.x;
    const int lane = tid & 31;
    const int wave = tid >> 5;
    const int wm = (wave & 1) * 64;
    const int wn = (wave >> 1) * 32;
    const int m0 = blockIdx.y * BM;
    const int n0 = blockIdx.x * BN;
    const int lr = lane & 15;
    const int kh = lane >> 4;

    v8f acc[4][2];
#pragma unroll
    for (int i = 0; i < 4; i++)
#pragma unroll
        for (int j = 0; j < 2; j++)
#pragma unroll
            for (int e = 0; e < 8; e++) acc[i][j][e] = 0.f;

    const int lrow = tid >> 1;
    const int lcol = (tid & 1) * 32;

    for (int k0 = 0; k0 < K; k0 += BK) {
        {   // A tile: fp32 -> bf16 convert while staging
            const float4* s4 = (const float4*)(A + (size_t)(m0 + lrow) * lda + k0 + lcol);
            v8bf* dst = (v8bf*)&As[lrow * LDSS + lcol];
#pragma unroll
            for (int e = 0; e < 4; e++) {
                float4 a = s4[2 * e], b = s4[2 * e + 1];
                v8bf v;
                v[0] = (bf16)a.x; v[1] = (bf16)a.y; v[2] = (bf16)a.z; v[3] = (bf16)a.w;
                v[4] = (bf16)b.x; v[5] = (bf16)b.y; v[6] = (bf16)b.z; v[7] = (bf16)b.w;
                dst[e] = v;
            }
        }
        {   // W tile (bf16, contiguous K)
            const v8bf* s8 = (const v8bf*)(Wt + (size_t)(n0 + lrow) * K + k0 + lcol);
            v8bf* dst = (v8bf*)&Bs[lrow * LDSS + lcol];
#pragma unroll
            for (int e = 0; e < 4; e++) dst[e] = s8[e];
        }
        __syncthreads();
        wmma_block(As, Bs, wm, wn, lr, kh, acc);
        __syncthreads();
    }
    gemm_epilogue<OT, SILU>(acc, bias, out, ldo, m0, n0, wm, wn, lr, kh);
}

// ---------------- bf16-input WMMA GEMM with TDM async staging --------------
// Double-buffered: wave 0 issues two TENSOR_LOAD_TO_LDS (A tile + W tile) for
// the next K-block, waits TENSORcnt<=2 to retire the current pair (in-order
// completion), then all 8 waves run 16 WMMAs on the ready buffer.
template <typename OT, bool SILU>
__global__ __launch_bounds__(256)
void k_gemm_tdm(const bf16* __restrict__ A, int lda,
                const bf16* __restrict__ Wt,
                const float* __restrict__ bias,
                OT* __restrict__ out, int ldo, int K) {
    constexpr int BM = 128, BN = 128, BK = 64, LDSS = BK + 8;
    __shared__ __align__(16) bf16 As[2][BM * LDSS];
    __shared__ __align__(16) bf16 Bs[2][BN * LDSS];

    const int tid  = threadIdx.x;
    const int lane = tid & 31;
    const int wave = tid >> 5;
    const int wm = (wave & 1) * 64;
    const int wn = (wave >> 1) * 32;
    const int m0 = blockIdx.y * BM;
    const int n0 = blockIdx.x * BN;
    const int lr = lane & 15;
    const int kh = lane >> 4;
    const int nIter = K / BK;

    v8f acc[4][2];
#pragma unroll
    for (int i = 0; i < 4; i++)
#pragma unroll
        for (int j = 0; j < 2; j++)
#pragma unroll
            for (int e = 0; e < 8; e++) acc[i][j][e] = 0.f;

    auto issue = [&](int it) {
        const int buf = it & 1;
        const int k0  = it * BK;
        tdm_load_2d((unsigned int)(uintptr_t)&As[buf][0],
                    A + (size_t)m0 * lda + k0, BM, BK, (unsigned)lda);
        tdm_load_2d((unsigned int)(uintptr_t)&Bs[buf][0],
                    Wt + (size_t)n0 * K + k0, BN, BK, (unsigned)K);
    };

    if (tid < 32) issue(0);   // prologue: fill buffer 0

    for (int it = 0; it < nIter; ++it) {
        if (tid < 32) {
            if (it + 1 < nIter) {
                issue(it + 1);                       // prefetch next tile pair
                __builtin_amdgcn_s_wait_tensorcnt(2); // current pair complete
            } else {
                __builtin_amdgcn_s_wait_tensorcnt(0);
            }
        }
        __syncthreads();
        wmma_block(As[it & 1], Bs[it & 1], wm, wn, lr, kh, acc);
        __syncthreads();
    }
    gemm_epilogue<OT, SILU>(acc, bias, out, ldo, m0, n0, wm, wn, lr, kh);
}

// ---------------- per-group attention (one wave per group of 32) ----------
__global__ __launch_bounds__(32)
void k_attention(const bf16* __restrict__ scs, const bf16* __restrict__ scv,
                 const bf16* __restrict__ sas, const bf16* __restrict__ sav,
                 bf16* __restrict__ pool_s, bf16* __restrict__ pool_v) {
    __shared__ __align__(16) float Sc[32][33];
    __shared__ float colw[32];
    const int b    = blockIdx.x;
    const int lane = threadIdx.x;
    const int lr = lane & 15, kh = lane >> 4;
    const size_t g0 = (size_t)b * GRP;

    v8f acc[2][2];
#pragma unroll
    for (int q = 0; q < 2; q++)
#pragma unroll
        for (int k = 0; k < 2; k++)
#pragma unroll
            for (int e = 0; e < 8; e++) acc[q][k][e] = 0.f;

    for (int kb = 0; kb < SD; kb += 32) {
        v16bf f[2];
#pragma unroll
        for (int t = 0; t < 2; t++)
            f[t] = mk_frag(scs + (g0 + t * 16 + lr) * SD + kb + kh * 8);
#pragma unroll
        for (int q = 0; q < 2; q++)
#pragma unroll
            for (int k = 0; k < 2; k++)
                acc[q][k] = __builtin_amdgcn_wmma_f32_16x16x32_bf16(
                    false, f[q], false, f[k], (short)0, acc[q][k], false, false);
    }
    for (int kb = 0; kb < VD; kb += 32) {
        v16bf f[2];
#pragma unroll
        for (int t = 0; t < 2; t++)
            f[t] = mk_frag(scv + (g0 + t * 16 + lr) * VD + kb + kh * 8);
#pragma unroll
        for (int q = 0; q < 2; q++)
#pragma unroll
            for (int k = 0; k < 2; k++)
                acc[q][k] = __builtin_amdgcn_wmma_f32_16x16x32_bf16(
                    false, f[q], false, f[k], (short)0, acc[q][k], false, false);
    }

#pragma unroll
    for (int q = 0; q < 2; q++)
#pragma unroll
        for (int k = 0; k < 2; k++)
#pragma unroll
            for (int r = 0; r < 8; r++)
                Sc[q * 16 + kh * 8 + r][k * 16 + lr] = acc[q][k][r];
    __syncthreads();

    {   // softmax, one row per lane
        float mx = -3.4e38f;
        for (int k = 0; k < 32; k++) mx = fmaxf(mx, Sc[lane][k]);
        float sum = 0.f;
        for (int k = 0; k < 32; k++) {
            float e = __expf(Sc[lane][k] - mx);
            Sc[lane][k] = e;
            sum += e;
        }
        float inv = 1.f / sum;
        for (int k = 0; k < 32; k++) Sc[lane][k] *= inv;
    }
    __syncthreads();
    {   // column weights (sum over q), one column per lane
        float s = 0.f;
        for (int q = 0; q < 32; q++) s += Sc[q][lane];
        colw[lane] = s;
    }
    __syncthreads();

    for (int d = lane; d < SD; d += 32) {
        float s = 0.f;
        for (int k = 0; k < 32; k++) s += colw[k] * (float)sas[(g0 + k) * SD + d];
        pool_s[(size_t)b * SD + d] = (bf16)s;
    }
    for (int d = lane; d < VD; d += 32) {
        float s = 0.f;
        for (int k = 0; k < 32; k++) s += colw[k] * (float)sav[(g0 + k) * VD + d];
        pool_v[(size_t)b * VD + d] = (bf16)s;
    }
}

// ===========================================================================
extern "C" void kernel_launch(void* const* d_in, const int* in_sizes, int n_in,
                              void* d_out, int out_size, void* d_ws, size_t ws_size,
                              hipStream_t stream) {
    (void)in_sizes; (void)n_in; (void)out_size; (void)ws_size;

    const float* emb = (const float*)d_in[0];
    // d_in[1] = batch_indices (implied by layout, unused)
    const float* p[38];
    for (int i = 2; i < 38; i++) p[i] = (const float*)d_in[i];
    // param leaf order (dict insertion order):
    //  2..5  ds_phi_s  w1,b1,w2,b2        6..9  ds_phi_v
    // 10..13 ds_rho_s                    14..17 ds_rho_v
    // 18..21 sa_phi_s                    22..25 sa_phi_v
    // 26..29 sa_rho_s                    30..33 sa_rho_v
    // 34,35  att_s w,b                   36,37  att_v w,b

    float* out    = (float*)d_out;
    float* o_mp_s = out;                 // 2048*256
    float* o_mp_v = out + 524288;        // 2048*768
    float* o_ds_s = out + 2097152;       // 2048*256
    float* o_ds_v = out + 2621440;       // 2048*768
    float* o_sa_s = out + 4194304;       // 2048*256
    float* o_sa_v = out + 4718592;       // 2048*768

    char* ws = (char*)d_ws;
    size_t off = 0;
    auto alloc = [&](size_t bytes) -> void* {
        void* r = ws + off;
        off += (bytes + 255) & ~(size_t)255;
        return r;
    };

    // bf16 transposed weights
    bf16* wt[38] = {};
    auto wt_make = [&](int widx, int K, int Nc) {
        wt[widx] = (bf16*)alloc((size_t)K * Nc * sizeof(bf16));
        int total = K * Nc;
        k_conv_wt<<<(total + 255) / 256, 256, 0, stream>>>(p[widx], wt[widx], K, Nc);
    };
    const int sIdx[9] = {2, 4, 10, 12, 18, 20, 26, 28, 34};
    const int vIdx[9] = {6, 8, 14, 16, 22, 24, 30, 32, 36};
    for (int i = 0; i < 9; i++) wt_make(sIdx[i], SD, SD);
    for (int i = 0; i < 9; i++) wt_make(vIdx[i], VD, VD);

    // activation workspace (bf16)
    bf16* HID    = (bf16*)alloc((size_t)NTOK * VD * 2);  // MLP hidden (reused)
    bf16* TOK    = (bf16*)alloc((size_t)NTOK * VD * 2);  // ds tok out -> later sa_v
    bf16* SAS    = (bf16*)alloc((size_t)NTOK * SD * 2);  // sa_s
    bf16* SCS    = (bf16*)alloc((size_t)NTOK * SD * 2);  // sc_s
    bf16* SCV    = (bf16*)alloc((size_t)NTOK * VD * 2);  // sc_v
    bf16* P_DS_S = (bf16*)alloc((size_t)NB * SD * 2);
    bf16* P_DS_V = (bf16*)alloc((size_t)NB * VD * 2);
    bf16* P_SA_S = (bf16*)alloc((size_t)NB * SD * 2);
    bf16* P_SA_V = (bf16*)alloc((size_t)NB * VD * 2);
    bf16* PHID   = (bf16*)alloc((size_t)NB * VD * 2);

    // mp_s / mp_v (exact fp32)
    k_meanpool_input<<<(NB * DD + 255) / 256, 256, 0, stream>>>(emb, o_mp_s, o_mp_v);

    auto G = [](int M, int Nc) { return dim3((unsigned)(Nc / 128), (unsigned)(M / 128)); };

    // ---- ds branch, scalar ----
    k_gemm_f32<bf16, true><<<G(NTOK, SD), 256, 0, stream>>>(emb, DD, wt[2], p[3], HID, SD, SD);
    k_gemm_tdm<bf16, true><<<G(NTOK, SD), 256, 0, stream>>>(HID, SD, wt[4], p[5], TOK, SD, SD);
    k_pool_bf16<<<(NB * SD + 255) / 256, 256, 0, stream>>>(TOK, P_DS_S, SD);
    // ---- ds branch, vector ----
    k_gemm_f32<bf16, true><<<G(NTOK, VD), 256, 0, stream>>>(emb + SD, DD, wt[6], p[7], HID, VD, VD);
    k_gemm_tdm<bf16, true><<<G(NTOK, VD), 256, 0, stream>>>(HID, VD, wt[8], p[9], TOK, VD, VD);
    k_pool_bf16<<<(NB * VD + 255) / 256, 256, 0, stream>>>(TOK, P_DS_V, VD);
    // ---- ds rho ----
    k_gemm_tdm<bf16, true><<<G(NB, SD), 256, 0, stream>>>(P_DS_S, SD, wt[10], p[11], PHID, SD, SD);
    k_gemm_tdm<float, true><<<G(NB, SD), 256, 0, stream>>>(PHID, SD, wt[12], p[13], o_ds_s, SD, SD);
    k_gemm_tdm<bf16, true><<<G(NB, VD), 256, 0, stream>>>(P_DS_V, VD, wt[14], p[15], PHID, VD, VD);
    k_gemm_tdm<float, true><<<G(NB, VD), 256, 0, stream>>>(PHID, VD, wt[16], p[17], o_ds_v, VD, VD);
    // ---- sa phi ----
    k_gemm_f32<bf16, true><<<G(NTOK, SD), 256, 0, stream>>>(emb, DD, wt[18], p[19], HID, SD, SD);
    k_gemm_tdm<bf16, true><<<G(NTOK, SD), 256, 0, stream>>>(HID, SD, wt[20], p[21], SAS, SD, SD);
    k_gemm_f32<bf16, true><<<G(NTOK, VD), 256, 0, stream>>>(emb + SD, DD, wt[22], p[23], HID, VD, VD);
    k_gemm_tdm<bf16, true><<<G(NTOK, VD), 256, 0, stream>>>(HID, VD, wt[24], p[25], TOK, VD, VD);  // TOK = sa_v
    // ---- att projections (linear) ----
    k_gemm_tdm<bf16, false><<<G(NTOK, SD), 256, 0, stream>>>(SAS, SD, wt[34], p[35], SCS, SD, SD);
    k_gemm_tdm<bf16, false><<<G(NTOK, VD), 256, 0, stream>>>(TOK, VD, wt[36], p[37], SCV, VD, VD);
    // ---- group attention ----
    k_attention<<<NB, 32, 0, stream>>>(SCS, SCV, SAS, TOK, P_SA_S, P_SA_V);
    // ---- sa rho ----
    k_gemm_tdm<bf16, true><<<G(NB, SD), 256, 0, stream>>>(P_SA_S, SD, wt[26], p[27], PHID, SD, SD);
    k_gemm_tdm<float, true><<<G(NB, SD), 256, 0, stream>>>(PHID, SD, wt[28], p[29], o_sa_s, SD, SD);
    k_gemm_tdm<bf16, true><<<G(NB, VD), 256, 0, stream>>>(P_SA_V, VD, wt[30], p[31], PHID, VD, VD);
    k_gemm_tdm<float, true><<<G(NB, VD), 256, 0, stream>>>(PHID, VD, wt[32], p[33], o_sa_v, VD, VD);
}
